// SimpleDyRAggregator_57612691309383
// MI455X (gfx1250) — matrix operand order repaired and checked by reading, not compiled, not run on hardware
//
#include <hip/hip_runtime.h>
#include <hip/hip_bf16.h>

typedef float   fl4   __attribute__((ext_vector_type(4)));
typedef float   v8f   __attribute__((ext_vector_type(8)));
typedef __bf16  v16bf __attribute__((ext_vector_type(16)));

typedef unsigned int   u32;
typedef unsigned short u16;

#define N_CAPS   200
#define N_PAD    208            // 13 * 16
#define NTILE    13
#define DIM      256
#define UST      264            // bf16 row stride: 528B = 33*16B -> aligned + conflict-free
#define GS       211            // f32 row stride for G: odd -> conflict-free row reads
#define NTHREADS 512
#define NWAVES   16

// LDS layout (bytes)
#define OFF_U    0
#define SZ_U     (N_PAD * UST * 2)       // 109824
#define OFF_G    (OFF_U + SZ_U)
#define SZ_G     (N_PAD * GS * 4)        // 175552
#define OFF_B    (OFF_G + SZ_G)
#define OFF_C    (OFF_B + N_PAD * 4)
#define OFF_Y    (OFF_C + N_PAD * 4)
#define OFF_RED  (OFF_Y + N_PAD * 4)
#define SMEM_TOTAL (OFF_RED + 64 * 4)    // 288128 B  (< 320KB per WGP)

union FragU { fl4 f[2]; v16bf v; };      // 32B reinterpret: two b128 LDS loads -> 16 bf16

__device__ __forceinline__ u16 f2bf(float f) {           // round-to-nearest-even f32 -> bf16
    u32 x = __float_as_uint(f);
    x += 0x7FFFu + ((x >> 16) & 1u);
    return (u16)(x >> 16);
}
__device__ __forceinline__ float bf2f(u16 h) {
    return __uint_as_float(((u32)h) << 16);
}

__device__ __forceinline__ float waveRedSum(float v) {
#pragma unroll
    for (int o = 16; o > 0; o >>= 1) v += __shfl_xor(v, o, 32);
    return v;
}
__device__ __forceinline__ float waveRedMax(float v) {
#pragma unroll
    for (int o = 16; o > 0; o >>= 1) v = fmaxf(v, __shfl_xor(v, o, 32));
    return v;
}
__device__ __forceinline__ float blockRedSum(float v, float* red) {
    const int w = threadIdx.x >> 5, l = threadIdx.x & 31;
    v = waveRedSum(v);
    __syncthreads();
    if (l == 0) red[w] = v;
    __syncthreads();
    float t = (l < NWAVES) ? red[l] : 0.0f;   // every wave folds the partials -> no broadcast
    return waveRedSum(t);
}
__device__ __forceinline__ float blockRedMax(float v, float* red) {
    const int w = threadIdx.x >> 5, l = threadIdx.x & 31;
    v = waveRedMax(v);
    __syncthreads();
    if (l == 0) red[w] = v;
    __syncthreads();
    float t = (l < NWAVES) ? red[l] : -3.402823466e38f;
    return waveRedMax(t);
}

extern "C" __global__ __launch_bounds__(NTHREADS, 1)
void caps_routing_kernel(const float* __restrict__ embeds,
                         const float* __restrict__ weights,
                         float* __restrict__ out, int B) {
    extern __shared__ char smem[];
    u16*   U    = (u16*)  (smem + OFF_U);
    float* G    = (float*)(smem + OFF_G);
    float* bArr = (float*)(smem + OFF_B);
    float* cArr = (float*)(smem + OFF_C);
    float* yArr = (float*)(smem + OFF_Y);
    float* red  = (float*)(smem + OFF_RED);

    const int tid  = threadIdx.x;
    const int wave = tid >> 5;
    const int lane = tid & 31;
    const int l15  = lane & 15;
    const int kh   = lane >> 4;                      // 0: lanes 0-15, 1: lanes 16-31
    const int b    = blockIdx.x;

    // ---------------- Phase 1: load embeds[b], row-normalize, pack bf16 U into LDS ------
    const float* ebase = embeds + (size_t)b * (N_CAPS * DIM);
    for (int r = wave; r < N_CAPS; r += NWAVES) {    // one wave per row (wave32 = 2 fl4/lane)
        const fl4* gp = (const fl4*)(ebase + r * DIM);
        fl4 c0 = gp[lane];
        fl4 c1 = gp[32 + lane];
        float ssq = c0.x*c0.x + c0.y*c0.y + c0.z*c0.z + c0.w*c0.w
                  + c1.x*c1.x + c1.y*c1.y + c1.z*c1.z + c1.w*c1.w;
        ssq = waveRedSum(ssq);
        float sc = 1.0f / fmaxf(sqrtf(ssq), 1e-12f);  // F.normalize eps
        u32* urow = (u32*)(U + r * UST);
        urow[lane * 2]          = (u32)f2bf(c0.x*sc) | ((u32)f2bf(c0.y*sc) << 16);
        urow[lane * 2 + 1]      = (u32)f2bf(c0.z*sc) | ((u32)f2bf(c0.w*sc) << 16);
        urow[64 + lane * 2]     = (u32)f2bf(c1.x*sc) | ((u32)f2bf(c1.y*sc) << 16);
        urow[64 + lane * 2 + 1] = (u32)f2bf(c1.z*sc) | ((u32)f2bf(c1.w*sc) << 16);
    }
    // zero pad rows 200..207 so padded G entries are exactly 0
    {
        u32* pz = (u32*)(U + N_CAPS * UST);
        for (int i = tid; i < (N_PAD - N_CAPS) * UST / 2; i += NTHREADS) pz[i] = 0u;
    }
    // init routing logits
    if (tid < N_PAD)
        bArr[tid] = (tid < N_CAPS) ? weights[(size_t)b * N_CAPS + tid] : -1e30f;
    __syncthreads();

    // ---------------- Phase 2: G = U * U^T via v_wmma_f32_16x16x32_bf16 -----------------
    // 13x13 tiles of 16x16, K = 256 = 8 steps of 32. B-operand = U^T, gathered straight
    // from row-major U per the ISA wave32 B layout (K contiguous per lane).
    for (int t = wave; t < NTILE * NTILE; t += NWAVES) {
        const int I = t / NTILE, J = t % NTILE;
        const u16* arow = U + (I * 16 + l15) * UST;
        const u16* brow = U + (J * 16 + l15) * UST;
        v8f acc = {};
#pragma unroll
        for (int ks = 0; ks < 8; ++ks) {
            const int k0 = ks * 32;
            FragU a, bf;
            // A (16x32 bf16): VGPR0-3 = K 0..7 (+8*kh), VGPR4-7 = K 16..23 (+8*kh)
            a.f[0] = *(const fl4*)(arow + k0 + 8 * kh);
            a.f[1] = *(const fl4*)(arow + k0 + 16 + 8 * kh);
            // B (32x16 bf16): VGPR0-3 = K 0..7 (+8*kh), VGPR4-7 = K 16..23 (+8*kh), N = lane&15 (row of U = col of B)
            bf.f[0] = *(const fl4*)(brow + k0 + 8 * kh);
            bf.f[1] = *(const fl4*)(brow + k0 + 16 + 8 * kh);
            acc = __builtin_amdgcn_wmma_f32_16x16x32_bf16(
                false, a.v, false, bf.v, (short)0, acc, false, false);
        }
        // C/D layout: VGPR v -> row I*16 + v + 8*kh, col J*16 + (lane&15)
        float* gout = G + (I * 16 + 8 * kh) * GS + J * 16 + l15;
#pragma unroll
        for (int v = 0; v < 8; ++v) gout[v * GS] = acc[v];
    }
    __syncthreads();

    // ---------------- Phase 3: routing iterations (all on-chip) ------------------------
    // y = G c gives both agreement (U v = scale * y) and ssn = c^T G c = s^T s.
    for (int it = 0; it < 3; ++it) {
        // c = softmax(b) * N
        float x  = (tid < N_CAPS) ? bArr[tid] : -3.402823466e38f;
        float mx = blockRedMax(x, red);
        float e  = (tid < N_CAPS) ? __expf(x - mx) : 0.0f;
        float se = blockRedSum(e, red);
        float cv = e * ((float)N_CAPS / se);
        if (tid < N_PAD) cArr[tid] = (tid < N_CAPS) ? cv : 0.0f;
        __syncthreads();

        if (it < 2) {
            if (tid < N_PAD) {
                const float* grow = G + tid * GS;   // odd stride -> conflict-free
                float y = 0.0f;
#pragma unroll 8
                for (int n = 0; n < N_PAD; ++n) y = fmaf(grow[n], cArr[n], y);
                yArr[tid] = y;
            }
            __syncthreads();
            float p     = (tid < N_CAPS) ? cArr[tid] * yArr[tid] : 0.0f;
            float ssn   = blockRedSum(p, red);
            float scale = ssn / (1.0f + ssn) * rsqrtf(ssn + 1e-9f);
            if (tid < N_CAPS) bArr[tid] += scale * yArr[tid];
            __syncthreads();
        } else {
            // final: s = U^T c (once), v = squash(s)
            float s = 0.0f;
            if (tid < DIM) {
                const u16* ucol = U + tid;
#pragma unroll 4
                for (int n = 0; n < N_CAPS; ++n)
                    s = fmaf(bf2f(ucol[n * UST]), cArr[n], s);
            }
            float p     = (tid < DIM) ? s * s : 0.0f;
            float ssn   = blockRedSum(p, red);
            float scale = ssn / (1.0f + ssn) * rsqrtf(ssn + 1e-9f);
            if (tid < DIM)    out[(size_t)b * DIM + tid] = scale * s;
            if (tid < N_CAPS) out[(size_t)B * DIM + (size_t)b * N_CAPS + tid] = cArr[tid];
        }
    }
}

extern "C" void kernel_launch(void* const* d_in, const int* in_sizes, int n_in,
                              void* d_out, int out_size, void* d_ws, size_t ws_size,
                              hipStream_t stream) {
    const float* embeds  = (const float*)d_in[0];
    const float* weights = (const float*)d_in[1];
    float* out = (float*)d_out;
    const int B = in_sizes[1] / N_CAPS;   // weights is [B, N]
    caps_routing_kernel<<<dim3(B), dim3(NTHREADS), SMEM_TOTAL, stream>>>(
        embeds, weights, out, B);
}
